// ConformerLayer_85830626443311
// MI455X (gfx1250) — matrix operand order repaired and checked by reading, not compiled, not run on hardware
//
#include <hip/hip_runtime.h>
#include <hip/hip_bf16.h>
#include <math.h>

#define B_ 2
#define N_ 2048
#define CACHE_ 512
#define KV_ 2560
#define D_ 768
#define H_ 12
#define HD_ 64
#define FF_ 3072
#define T_ (B_*N_)
#define KSZ_ 31
#define EPS_ 1e-5f

// CDNA5 async global->LDS copies (ASYNCcnt-tracked, bypass VGPRs)
#define USE_ASYNC_LDS 1

typedef __attribute__((ext_vector_type(16))) __bf16 v16bf;
typedef __attribute__((ext_vector_type(8)))  float  v8f;

#define DEVFN static __device__ __forceinline__

DEVFN unsigned short f32_to_bf16(float f) {
  unsigned int u = __builtin_bit_cast(unsigned int, f);
  u = (u + 0x7FFFu + ((u >> 16) & 1u)) >> 16;
  return (unsigned short)u;
}

// 16B global -> LDS copy. Flat LDS address truncates to the LDS byte offset
// (ISA section 10.2: LDS offset is the low 32 address bits).
DEVFN void async_copy_b128(const void* g, void* l) {
#if USE_ASYNC_LDS
  unsigned lds = (unsigned)(size_t)l;
  unsigned long long ga = (unsigned long long)(size_t)g;
  asm volatile("global_load_async_to_lds_b128 %0, %1, off"
               :: "v"(lds), "v"(ga) : "memory");
#else
  *(uint4*)l = *(const uint4*)g;
#endif
}
DEVFN void async_wait() {
#if USE_ASYNC_LDS
  asm volatile("s_wait_asynccnt 0x0" ::: "memory");
#endif
}

struct U32x8 { uint4 lo, hi; };
// Build a v16bf fragment from two 16-byte LDS chunks (2x ds_load_b128).
DEVFN v16bf frag_2x8(const unsigned short* p0, const unsigned short* p1) {
  U32x8 t;
  t.lo = *(const uint4*)p0;
  t.hi = *(const uint4*)p1;
  return __builtin_bit_cast(v16bf, t);
}
// A fragment (16x32 MxK, row-major tile): lane row = rowbase + lane%16,
// chunks at k = kb and kb+16, kb = kstep + (lane<16 ? 0 : 8).
DEVFN v16bf load_a_frag(const unsigned short* S, int ld, int rowbase, int kstep, int lane) {
  int row = rowbase + (lane & 15);
  int kb  = kstep + ((lane < 16) ? 0 : 8);
  const unsigned short* p = S + row * ld + kb;
  return frag_2x8(p, p + 16);
}
// B fragment from K-contiguous tile Bt[n][k]: lane col = colbase + lane%16,
// chunks at k = kb and kb+16, kb = kstep + (lane<16 ? 0 : 8).
DEVFN v16bf load_b_frag_kc(const unsigned short* S, int ld, int colbase, int kstep, int lane) {
  int col = colbase + (lane & 15);
  int kb  = kstep + ((lane < 16) ? 0 : 8);
  const unsigned short* p = S + col * ld + kb;
  return frag_2x8(p, p + 16);
}

// =========================================================================
// bf16 WMMA GEMM, double-buffered with async LDS copies.
// out = alpha*act(A@W + bias) + residual
// A: [M x K] bf16 row-major. Wt: [Nc x K] bf16 (TRANSPOSED). act: 0=none,1=gelu.
// Tile: 128(M) x 64(N) x 64(K); 8 waves each own 16x64.
// =========================================================================
#define GTM 128
#define GTN 64
#define GTK 64
__global__ __launch_bounds__(256) void gemm_bf16_kernel(
    const unsigned short* __restrict__ A, const unsigned short* __restrict__ Wt,
    const float* __restrict__ bias, const float* __restrict__ residual,
    float* __restrict__ outf, unsigned short* __restrict__ outb,
    int M, int K, int Nc, float alpha, int act) {
  __shared__ unsigned short As[2][GTM][GTK + 8];   // row-major MxK
  __shared__ unsigned short Bs[2][GTN][GTK + 8];   // K-contiguous per column

  const int m0   = blockIdx.y * GTM;
  const int n0   = blockIdx.x * GTN;
  const int tid  = threadIdx.x;
  const int lane = tid & 31;
  const int wave = tid >> 5;
  const int wm   = wave * 16;

  v8f acc[4] = {};

  auto stage = [&](int buf, int k0) {
#pragma unroll
    for (int i = 0; i < 4; ++i) {  // A tile: 128 rows x 8 chunks
      int s = tid + i * 256;
      int r = s >> 3, cg = (s & 7) * 8;
      async_copy_b128(A + (size_t)(m0 + r) * K + k0 + cg, &As[buf][r][cg]);
    }
#pragma unroll
    for (int i = 0; i < 2; ++i) {  // B tile: 64 rows x 8 chunks
      int s = tid + i * 256;
      int r = s >> 3, cg = (s & 7) * 8;
      async_copy_b128(Wt + (size_t)(n0 + r) * K + k0 + cg, &Bs[buf][r][cg]);
    }
  };

  stage(0, 0);
  async_wait();
  __syncthreads();

  const int nk = K / GTK;
  for (int kt = 0; kt < nk; ++kt) {
    const int cur = kt & 1;
    if (kt + 1 < nk) stage(1 - cur, (kt + 1) * GTK);  // prefetch next tile
    const unsigned short* Ab = &As[cur][0][0];
    const unsigned short* Bb = &Bs[cur][0][0];
#pragma unroll
    for (int ks = 0; ks < GTK; ks += 32) {
      // preload all fragments, then issue the 4 WMMAs back-to-back
      v16bf af = load_a_frag(Ab, GTK + 8, wm, ks, lane);
      v16bf bf[4];
#pragma unroll
      for (int t = 0; t < 4; ++t) bf[t] = load_b_frag_kc(Bb, GTK + 8, t * 16, ks, lane);
#pragma unroll
      for (int t = 0; t < 4; ++t)
        acc[t] = __builtin_amdgcn_wmma_f32_16x16x32_bf16(
            false, af, false, bf[t], (short)0, acc[t], false, false);
    }
    async_wait();
    __syncthreads();
  }

  // epilogue: lane l, vgpr r : row = r + 8*(l>=16), col = t*16 + l%16
  const int hb   = lane >> 4;
  const int nloc = lane & 15;
#pragma unroll
  for (int t = 0; t < 4; ++t) {
    int col  = n0 + t * 16 + nloc;
    float bv = bias ? bias[col] : 0.0f;
#pragma unroll
    for (int r = 0; r < 8; ++r) {
      int row = m0 + wm + r + hb * 8;
      float v = acc[t][r] + bv;
      if (act == 1) {  // tanh-approx gelu
        float c = v + 0.044715f * v * v * v;
        v = 0.5f * v * (1.0f + tanhf(0.7978845608028654f * c));
      }
      v *= alpha;
      size_t idx = (size_t)row * Nc + col;
      if (residual) v += residual[idx];
      if (outf) outf[idx] = v;
      if (outb) outb[idx] = f32_to_bf16(v);
    }
  }
}

// =========================================================================
// RMSNorm over D columns of one row; optional pad zeroing; f32 and/or bf16 out
// =========================================================================
__global__ __launch_bounds__(256) void rmsnorm_kernel(
    const float* __restrict__ x, const float* __restrict__ w,
    const unsigned char* __restrict__ pad,
    float* __restrict__ outf, unsigned short* __restrict__ outb, int D) {
  int row = blockIdx.x;
  const float* xr = x + (size_t)row * D;
  __shared__ float red[8];
  float s = 0.f;
  for (int i = threadIdx.x; i < D; i += 256) { float v = xr[i]; s += v * v; }
  for (int off = 16; off > 0; off >>= 1) s += __shfl_xor(s, off, 32);
  if ((threadIdx.x & 31) == 0) red[threadIdx.x >> 5] = s;
  __syncthreads();
  if (threadIdx.x < 8) {
    float v = red[threadIdx.x];
    for (int off = 4; off > 0; off >>= 1) v += __shfl_xor(v, off, 8);
    if (threadIdx.x == 0) red[0] = v;
  }
  __syncthreads();
  float inv = rsqrtf(red[0] / (float)D + EPS_);
  float z = (pad && pad[row]) ? 0.0f : 1.0f;
  for (int i = threadIdx.x; i < D; i += 256) {
    float v = xr[i] * inv * w[i] * z;
    if (outf) outf[(size_t)row * D + i] = v;
    if (outb) outb[(size_t)row * D + i] = f32_to_bf16(v);
  }
}

// =========================================================================
// QKV postprocess: de-interleave (h*192 + d*3 + {q,k,v}), qk rmsnorm, RoPE,
// emit kv-cache (pre-rope), q/k in [B][H][pos][64], V TRANSPOSED [B][H][64][KV].
// =========================================================================
__global__ __launch_bounds__(64) void qkv_post_kernel(
    const float* __restrict__ qkv, const float* __restrict__ qnw,
    const float* __restrict__ knw, unsigned short* __restrict__ qbf,
    unsigned short* __restrict__ kbf, unsigned short* __restrict__ vbf,
    float* __restrict__ kvout) {
  int t = blockIdx.x, h = blockIdx.y, d = threadIdx.x;
  int b = t / N_, n = t % N_;
  const float* base = qkv + (size_t)t * (3 * D_) + h * 192 + d * 3;
  float q = base[0], k = base[1], v = base[2];
  __shared__ float red[2][2];
  __shared__ float qs[64], ks[64];
  float sq = q * q, sk = k * k;
  for (int off = 16; off > 0; off >>= 1) {
    sq += __shfl_xor(sq, off, 32);
    sk += __shfl_xor(sk, off, 32);
  }
  if ((d & 31) == 0) { red[0][d >> 5] = sq; red[1][d >> 5] = sk; }
  __syncthreads();
  q *= rsqrtf((red[0][0] + red[0][1]) / 64.f + EPS_) * qnw[d];
  k *= rsqrtf((red[1][0] + red[1][1]) / 64.f + EPS_) * knw[d];
  int p = CACHE_ + n;
  kvout[((((size_t)b * KV_ + p) * 2 + 0) * H_ + h) * 64 + d] = k;  // pre-rope
  kvout[((((size_t)b * KV_ + p) * 2 + 1) * H_ + h) * 64 + d] = v;
  qs[d] = q; ks[d] = k;
  __syncthreads();
  int dh = d & 31;
  float ang = (float)p * __powf(10000.f, -(float)dh / 32.f);
  float c = __cosf(ang), sn = __sinf(ang);
  float qr = (d < 32) ? -qs[d + 32] : qs[d - 32];
  float kr = (d < 32) ? -ks[d + 32] : ks[d - 32];
  qbf[(((size_t)b * H_ + h) * N_ + n) * 64 + d]  = f32_to_bf16(q * c + qr * sn);
  kbf[(((size_t)b * H_ + h) * KV_ + p) * 64 + d] = f32_to_bf16(k * c + kr * sn);
  vbf[(((size_t)b * H_ + h) * 64 + d) * KV_ + p] = f32_to_bf16(v);  // transposed
}

__global__ __launch_bounds__(64) void cached_kv_kernel(
    const float* __restrict__ cached, unsigned short* __restrict__ kbf,
    unsigned short* __restrict__ vbf, float* __restrict__ kvout) {
  int t = blockIdx.x, h = blockIdx.y, d = threadIdx.x;
  int b = t / CACHE_, c = t % CACHE_;
  float k = cached[((((size_t)b * CACHE_ + c) * 2 + 0) * H_ + h) * 64 + d];
  float v = cached[((((size_t)b * CACHE_ + c) * 2 + 1) * H_ + h) * 64 + d];
  kvout[((((size_t)b * KV_ + c) * 2 + 0) * H_ + h) * 64 + d] = k;
  kvout[((((size_t)b * KV_ + c) * 2 + 1) * H_ + h) * 64 + d] = v;
  __shared__ float ks[64];
  ks[d] = k;
  __syncthreads();
  int dh = d & 31;
  float ang = (float)c * __powf(10000.f, -(float)dh / 32.f);
  float kr = (d < 32) ? -ks[d + 32] : ks[d - 32];
  kbf[(((size_t)b * H_ + h) * KV_ + c) * 64 + d] =
      f32_to_bf16(k * __cosf(ang) + kr * __sinf(ang));
  vbf[(((size_t)b * H_ + h) * 64 + d) * KV_ + c] = f32_to_bf16(v);  // transposed
}

// =========================================================================
// Flash attention, WMMA, async K/V staging. Block = 8 waves, Q tile 128 rows,
// K/V tiles of 64. Ks [kv][dim] (K-contig for QK^T B-frags); Vs [dim][kv]
// (K-contig for AV B-frags). Online softmax stats per-lane.
// =========================================================================
__global__ __launch_bounds__(256) void flash_attn_kernel(
    const unsigned short* __restrict__ qbf, const unsigned short* __restrict__ kbf,
    const unsigned short* __restrict__ vbf, const unsigned char* __restrict__ amask,
    const unsigned char* __restrict__ pad, unsigned short* __restrict__ outb) {
  __shared__ unsigned short Qs[128][72];
  __shared__ unsigned short Ks[64][72];   // [kv][dim]
  __shared__ unsigned short Vs[64][72];   // [dim][kv]
  __shared__ unsigned short Ps[8][16][72];
  __shared__ unsigned char  msk[64];

  const int qb = blockIdx.x, h = blockIdx.y, b = blockIdx.z;
  const int tid = threadIdx.x, lane = tid & 31, wave = tid >> 5;
  const int q0 = qb * 128;
  const int hb = lane >> 4, nloc = lane & 15;

#pragma unroll
  for (int i = 0; i < 4; ++i) {   // Q tile: 1024 chunks, async
    int s = tid + i * 256;
    int r = s >> 3, cg = (s & 7) * 8;
    async_copy_b128(qbf + (((size_t)(b * H_ + h) * N_) + q0 + r) * 64 + cg,
                    &Qs[r][cg]);
  }

  v8f oacc[4] = {};
  float m_i[8], l_i[8];
#pragma unroll
  for (int r = 0; r < 8; ++r) { m_i[r] = -1e30f; l_i[r] = 0.f; }

  for (int kt = 0; kt < KV_ / 64; ++kt) {
    __syncthreads();  // previous tile fully consumed
#pragma unroll
    for (int i = 0; i < 2; ++i) {  // K and V tiles: 512 chunks each, async
      int s = tid + i * 256;
      int r = s >> 3, cg = (s & 7) * 8;
      async_copy_b128(
          kbf + (((size_t)(b * H_ + h) * KV_) + kt * 64 + r) * 64 + cg, &Ks[r][cg]);
      async_copy_b128(
          vbf + ((size_t)(b * H_ + h) * 64 + r) * KV_ + kt * 64 + cg, &Vs[r][cg]);
    }
    if (tid < 64) msk[tid] = amask[(size_t)b * KV_ + kt * 64 + tid];
    async_wait();
    __syncthreads();

    // S = Q_w (16x64) @ K_tile^T : B(kdim,kv)=Ks[kv][kdim] is K-contiguous
    v8f S[4] = {};
#pragma unroll
    for (int ks = 0; ks < 64; ks += 32) {
      v16bf af = load_a_frag(&Qs[0][0], 72, wave * 16, ks, lane);
      v16bf bf[4];
#pragma unroll
      for (int t = 0; t < 4; ++t) bf[t] = load_b_frag_kc(&Ks[0][0], 72, t * 16, ks, lane);
#pragma unroll
      for (int t = 0; t < 4; ++t)
        S[t] = __builtin_amdgcn_wmma_f32_16x16x32_bf16(
            false, af, false, bf[t], (short)0, S[t], false, false);
    }
#pragma unroll
    for (int t = 0; t < 4; ++t) {
      bool ok = msk[t * 16 + nloc] != 0;
#pragma unroll
      for (int r = 0; r < 8; ++r) {
        float v = S[t][r] * 0.125f;  // 1/sqrt(64)
        S[t][r] = ok ? v : -1e30f;
      }
    }
#pragma unroll
    for (int r = 0; r < 8; ++r) {
      float mx = fmaxf(fmaxf(S[0][r], S[1][r]), fmaxf(S[2][r], S[3][r]));
      for (int off = 1; off < 16; off <<= 1) mx = fmaxf(mx, __shfl_xor(mx, off, 16));
      float mnew = fmaxf(m_i[r], mx);
      float al = __expf(m_i[r] - mnew);
      float rs = 0.f;
#pragma unroll
      for (int t = 0; t < 4; ++t) {
        float p = __expf(S[t][r] - mnew);
        S[t][r] = p;
        rs += p;
      }
      for (int off = 1; off < 16; off <<= 1) rs += __shfl_xor(rs, off, 16);
      l_i[r] = l_i[r] * al + rs;
      m_i[r] = mnew;
#pragma unroll
      for (int t = 0; t < 4; ++t) oacc[t][r] *= al;
    }
    // spill P (bf16) to per-wave LDS, reload as A fragments for P @ V
    unsigned short* pw = &Ps[wave][0][0];
#pragma unroll
    for (int t = 0; t < 4; ++t)
#pragma unroll
      for (int r = 0; r < 8; ++r)
        pw[(r + hb * 8) * 72 + t * 16 + nloc] = f32_to_bf16(S[t][r]);
#pragma unroll
    for (int ks = 0; ks < 64; ks += 32) {
      v16bf pa = load_a_frag(pw, 72, 0, ks, lane);
      v16bf vb[4];
#pragma unroll
      for (int t = 0; t < 4; ++t) vb[t] = load_b_frag_kc(&Vs[0][0], 72, t * 16, ks, lane);
#pragma unroll
      for (int t = 0; t < 4; ++t)
        oacc[t] = __builtin_amdgcn_wmma_f32_16x16x32_bf16(
            false, pa, false, vb[t], (short)0, oacc[t], false, false);
    }
  }
#pragma unroll
  for (int t = 0; t < 4; ++t)
#pragma unroll
    for (int r = 0; r < 8; ++r) {
      int n = q0 + wave * 16 + r + hb * 8;
      float o = oacc[t][r] / fmaxf(l_i[r], 1e-20f);
      if (pad && pad[b * N_ + n]) o = 0.f;
      outb[((size_t)(b * N_ + n)) * D_ + h * 64 + t * 16 + nloc] = f32_to_bf16(o);
    }
}

// ---- small elementwise / reduction kernels ------------------------------
// f32 [K][Nc] -> bf16 TRANSPOSED [Nc][K]
__global__ void f2bf_tr_kernel(const float* __restrict__ w,
                               unsigned short* __restrict__ o, int K, int Nc) {
  int i = blockIdx.x * 256 + threadIdx.x;
  if (i < K * Nc) {
    int k = i / Nc, n = i % Nc;
    o[(size_t)n * K + k] = f32_to_bf16(w[i]);
  }
}
__global__ void zero_kernel(float* p, int n) {
  int i = blockIdx.x * 256 + threadIdx.x;
  if (i < n) p[i] = 0.f;
}
__global__ void glu_kernel(const float* __restrict__ h, float* __restrict__ y) {
  size_t i = (size_t)blockIdx.x * 256 + threadIdx.x;
  size_t t = i / D_, c = i % D_;
  float a = h[t * (2 * D_) + c], g = h[t * (2 * D_) + D_ + c];
  y[i] = a * (1.f / (1.f + __expf(-g)));
}
__global__ __launch_bounds__(256) void dwconv_kernel(
    const float* __restrict__ y, const float* __restrict__ dww,
    const float* __restrict__ dwb, float* __restrict__ y2,
    float* __restrict__ stats) {
  int c = blockIdx.x;
  int t = blockIdx.y * 256 + threadIdx.x;
  int b = t / N_, n = t % N_;
  float acc = dwb[c];
#pragma unroll
  for (int j = 0; j < KSZ_; ++j) {
    int nn = n + j - KSZ_ / 2;
    float xv = (nn >= 0 && nn < N_) ? y[((size_t)b * N_ + nn) * D_ + c] : 0.f;
    acc = fmaf(xv, dww[c * KSZ_ + j], acc);
  }
  y2[(size_t)t * D_ + c] = acc;
  float s = acc, s2 = acc * acc;
  for (int off = 16; off > 0; off >>= 1) {
    s += __shfl_xor(s, off, 32);
    s2 += __shfl_xor(s2, off, 32);
  }
  __shared__ float rs[8], rs2[8];
  if ((threadIdx.x & 31) == 0) { rs[threadIdx.x >> 5] = s; rs2[threadIdx.x >> 5] = s2; }
  __syncthreads();
  if (threadIdx.x == 0) {
    float a0 = 0, a1 = 0;
    for (int i = 0; i < 8; ++i) { a0 += rs[i]; a1 += rs2[i]; }
    atomicAdd(&stats[c], a0);
    atomicAdd(&stats[D_ + c], a1);
  }
}
__global__ void bn_swish_kernel(const float* __restrict__ y2,
                                const float* __restrict__ stats,
                                const float* __restrict__ g,
                                const float* __restrict__ bb,
                                unsigned short* __restrict__ outb) {
  size_t i = (size_t)blockIdx.x * 256 + threadIdx.x;
  int c = (int)(i % D_);
  float mean = stats[c] / (float)T_;
  float var = stats[D_ + c] / (float)T_ - mean * mean;
  float v = (y2[i] - mean) * rsqrtf(var + EPS_) * g[c] + bb[c];
  v = v * (1.f / (1.f + __expf(-v)));
  outb[i] = f32_to_bf16(v);
}

// =========================================================================
static void gemm(hipStream_t s, const unsigned short* A, const unsigned short* Wt,
                 const float* bias, const float* res, float* outf,
                 unsigned short* outb, int M, int K, int Nc, float alpha, int act) {
  dim3 g(Nc / GTN, M / GTM);
  gemm_bf16_kernel<<<g, 256, 0, s>>>(A, Wt, bias, res, outf, outb, M, K, Nc, alpha, act);
}

extern "C" void kernel_launch(void* const* d_in, const int* in_sizes, int n_in,
                              void* d_out, int out_size, void* d_ws, size_t ws_size,
                              hipStream_t stream) {
  const float* x      = (const float*)d_in[0];
  const unsigned char* amask = (const unsigned char*)d_in[1];
  const unsigned char* pmask = (const unsigned char*)d_in[2];
  const float* cached = (const float*)d_in[4];
  const float* ff1_nw = (const float*)d_in[5];
  const float* ff1_w1 = (const float*)d_in[6];
  const float* ff1_b1 = (const float*)d_in[7];
  const float* ff1_w2 = (const float*)d_in[8];
  const float* ff1_b2 = (const float*)d_in[9];
  const float* att_nw = (const float*)d_in[10];
  const float* qkv_w  = (const float*)d_in[11];
  const float* out_w  = (const float*)d_in[12];
  const float* q_nw   = (const float*)d_in[13];
  const float* k_nw   = (const float*)d_in[14];
  const float* cnv_nw = (const float*)d_in[15];
  const float* pw1_w  = (const float*)d_in[16];
  const float* pw1_b  = (const float*)d_in[17];
  const float* dw_w   = (const float*)d_in[18];
  const float* dw_b   = (const float*)d_in[19];
  const float* bn_g   = (const float*)d_in[20];
  const float* bn_b   = (const float*)d_in[21];
  const float* pw2_w  = (const float*)d_in[22];
  const float* pw2_b  = (const float*)d_in[23];
  const float* ff2_nw = (const float*)d_in[24];
  const float* ff2_w1 = (const float*)d_in[25];
  const float* ff2_b1 = (const float*)d_in[26];
  const float* ff2_w2 = (const float*)d_in[27];
  const float* ff2_b2 = (const float*)d_in[28];
  const float* out_nw = (const float*)d_in[29];

  float* outx  = (float*)d_out;
  float* kvout = outx + (size_t)T_ * D_;

  char* w8 = (char*)d_ws;
  size_t off = 0;
  auto alloc = [&](size_t bytes) -> void* {
    void* p = w8 + off;
    off += (bytes + 255) & ~(size_t)255;
    return p;
  };
  float* xa    = (float*)alloc((size_t)T_ * D_ * 4);
  float* xb    = (float*)alloc((size_t)T_ * D_ * 4);
  float* hbuf  = (float*)alloc((size_t)T_ * 2304 * 4);
  float* ybuf  = (float*)alloc((size_t)T_ * D_ * 4);
  float* y2buf = (float*)alloc((size_t)T_ * D_ * 4);
  float* stats = (float*)alloc(2 * D_ * 4);
  unsigned short* abf = (unsigned short*)alloc((size_t)T_ * FF_ * 2);
  unsigned short* hbf = (unsigned short*)alloc((size_t)T_ * FF_ * 2);
  unsigned short* wbf = (unsigned short*)alloc((size_t)D_ * FF_ * 2);
  unsigned short* qbf = (unsigned short*)alloc((size_t)B_ * H_ * N_ * 64 * 2);
  unsigned short* kbf = (unsigned short*)alloc((size_t)B_ * H_ * KV_ * 64 * 2);
  unsigned short* vbf = (unsigned short*)alloc((size_t)B_ * H_ * KV_ * 64 * 2);

  auto f2bf = [&](const float* src, int K, int Nc) {
    f2bf_tr_kernel<<<(K * Nc + 255) / 256, 256, 0, stream>>>(src, wbf, K, Nc);
  };

  // ---- FF1:  xa = 0.5*(gelu(rmsnorm(x)@w1+b1)@w2+b2) + x
  rmsnorm_kernel<<<T_, 256, 0, stream>>>(x, ff1_nw, nullptr, nullptr, abf, D_);
  f2bf(ff1_w1, D_, FF_);
  gemm(stream, abf, wbf, ff1_b1, nullptr, nullptr, hbf, T_, D_, FF_, 1.0f, 1);
  f2bf(ff1_w2, FF_, D_);
  gemm(stream, hbf, wbf, ff1_b2, x, xa, nullptr, T_, FF_, D_, 0.5f, 0);

  // ---- Attention
  rmsnorm_kernel<<<T_, 256, 0, stream>>>(xa, att_nw, pmask, nullptr, abf, D_);
  f2bf(qkv_w, D_, 3 * D_);
  gemm(stream, abf, wbf, nullptr, nullptr, hbuf, nullptr, T_, D_, 3 * D_, 1.0f, 0);
  qkv_post_kernel<<<dim3(T_, H_), 64, 0, stream>>>(hbuf, q_nw, k_nw, qbf, kbf, vbf, kvout);
  cached_kv_kernel<<<dim3(B_ * CACHE_, H_), 64, 0, stream>>>(cached, kbf, vbf, kvout);
  flash_attn_kernel<<<dim3(N_ / 128, H_, B_), 256, 0, stream>>>(qbf, kbf, vbf, amask,
                                                                pmask, abf);
  f2bf(out_w, D_, D_);
  gemm(stream, abf, wbf, nullptr, xa, xb, nullptr, T_, D_, D_, 1.0f, 0);

  // ---- Conv module -> xa
  rmsnorm_kernel<<<T_, 256, 0, stream>>>(xb, cnv_nw, pmask, nullptr, abf, D_);
  f2bf(pw1_w, D_, 2 * D_);
  gemm(stream, abf, wbf, pw1_b, nullptr, hbuf, nullptr, T_, D_, 2 * D_, 1.0f, 0);
  glu_kernel<<<(T_ * D_) / 256, 256, 0, stream>>>(hbuf, ybuf);
  zero_kernel<<<(2 * D_ + 255) / 256, 256, 0, stream>>>(stats, 2 * D_);
  dwconv_kernel<<<dim3(D_, T_ / 256), 256, 0, stream>>>(ybuf, dw_w, dw_b, y2buf, stats);
  bn_swish_kernel<<<(T_ * D_) / 256, 256, 0, stream>>>(y2buf, stats, bn_g, bn_b, abf);
  f2bf(pw2_w, D_, D_);
  gemm(stream, abf, wbf, pw2_b, nullptr, xa, nullptr, T_, D_, D_, 1.0f, 0);

  // ---- FF2 -> xb
  rmsnorm_kernel<<<T_, 256, 0, stream>>>(xa, ff2_nw, nullptr, nullptr, abf, D_);
  f2bf(ff2_w1, D_, FF_);
  gemm(stream, abf, wbf, ff2_b1, nullptr, nullptr, hbf, T_, D_, FF_, 1.0f, 1);
  f2bf(ff2_w2, FF_, D_);
  gemm(stream, hbf, wbf, ff2_b2, xa, xb, nullptr, T_, FF_, D_, 0.5f, 0);

  // ---- final norm -> d_out
  rmsnorm_kernel<<<T_, 256, 0, stream>>>(xb, out_nw, nullptr, outx, nullptr, D_);
}
